// Gemma3MoEAttention_26800595927645
// MI455X (gfx1250) — matrix operand order repaired and checked by reading, not compiled, not run on hardware
//
#include <hip/hip_runtime.h>
#include <hip/hip_bf16.h>
#include <cstdint>

// Problem constants (match reference)
#define BB    4
#define TT    2048
#define DD    2048
#define NHEAD 8
#define NKVH  4
#define HDIM  256
#define QKVD  4096            // (NHEAD + 2*NKVH) * HDIM
#define KOFF  2048            // NHEAD*HDIM
#define VOFF  3072            // (NHEAD+NKVH)*HDIM
#define ROPE_LOG2_BASE 13.287712379549449f   // log2(10000)

typedef __attribute__((ext_vector_type(16))) __bf16       v16bf;
typedef __attribute__((ext_vector_type(8)))  float        v8f;
typedef __attribute__((ext_vector_type(8)))  unsigned int v8u;

__device__ __forceinline__ unsigned short f2bf(float x) {
  return (unsigned short)(__float_as_uint(x) >> 16);   // truncate to bf16
}
__device__ __forceinline__ unsigned int pk2(float a, float b) {
  return (unsigned int)f2bf(a) | ((unsigned int)f2bf(b) << 16);
}
__device__ __forceinline__ v8u mk8(uint4 a, uint4 b) {
  v8u r;
  r[0] = a.x; r[1] = a.y; r[2] = a.z; r[3] = a.w;
  r[4] = b.x; r[5] = b.y; r[6] = b.z; r[7] = b.w;
  return r;
}
__device__ __forceinline__ v8f wmma_bf16(const v8u& a, const v8u& b, const v8f& c) {
  return __builtin_amdgcn_wmma_f32_16x16x32_bf16(
      false, __builtin_bit_cast(v16bf, a),
      false, __builtin_bit_cast(v16bf, b),
      (short)0, c, false, false);
}

// DPP cross-lane move (stays within a 16-lane DPP row; no LDS, no dscnt).
template <int CTRL>
__device__ __forceinline__ float dpp_movf(float x) {
  return __builtin_bit_cast(
      float, __builtin_amdgcn_update_dpp(0, __builtin_bit_cast(int, x), CTRL,
                                         0xF, 0xF, true));
}
// Butterfly reductions over each 16-lane half (matches WMMA C-layout rows).
__device__ __forceinline__ float red16_max(float x) {
  x = fmaxf(x, dpp_movf<0xB1>(x));    // quad_perm [1,0,3,2]  (xor 1)
  x = fmaxf(x, dpp_movf<0x4E>(x));    // quad_perm [2,3,0,1]  (xor 2)
  x = fmaxf(x, dpp_movf<0x141>(x));   // row_half_mirror      (combine quads)
  x = fmaxf(x, dpp_movf<0x140>(x));   // row_mirror           (combine octets)
  return x;
}
__device__ __forceinline__ float red16_sum(float x) {
  x += dpp_movf<0xB1>(x);
  x += dpp_movf<0x4E>(x);
  x += dpp_movf<0x141>(x);
  x += dpp_movf<0x140>(x);
  return x;
}
// Fast tanh via hardware v_exp; input is finite (pre-mask logits / 50).
__device__ __forceinline__ float fast_tanh(float x) {
  x = fminf(fmaxf(x, -15.f), 15.f);
  float e2 = __expf(2.f * x);
  return (e2 - 1.f) / (e2 + 1.f);
}

// ---------------------------------------------------------------------------
// C = A[M x K] * B[K x N], fp32 in/out, bf16 WMMA compute.  N, K compile-time.
// Block: 128 threads (4 waves). Tile 64x64, K step 32.
// 3-phase double-buffered pipeline: global loads (regs) -> WMMA on current
// LDS buffer -> pack + ds_store to other buffer -> one barrier per step.
// ---------------------------------------------------------------------------
#define AS_STRIDE 20   // u32 per A row (16 used + pad), 80B = 16B-aligned
#define BS_STRIDE 20   // u32 per B col (16 used + pad)

template <int N, int K>
__global__ void __launch_bounds__(128)
gemm_kernel(const float* __restrict__ A, const float* __restrict__ Bw,
            float* __restrict__ C) {
  __shared__ __align__(16) unsigned int As[2][64 * AS_STRIDE]; // [row][kpair]
  __shared__ __align__(16) unsigned int Bs[2][64 * BS_STRIDE]; // [col][kpair]

  const int tid  = threadIdx.x;
  const int wv   = tid >> 5;
  const int lane = tid & 31;
  const int l15  = lane & 15;
  const int half = lane >> 4;

  const int row0 = blockIdx.y * 64;
  const int col0 = blockIdx.x * 64;

  float2 fa[8];          // staged A slices (in-flight across compute)
  float4 fb0[2], fb1[2]; // staged B row pairs

  auto load_phase = [&](int kc) {
#pragma unroll
    for (int i = 0; i < 8; ++i) {
      int p = i * 128 + tid;
      fa[i] = *(const float2*)(A + (size_t)(row0 + (p >> 4)) * K + kc +
                               (p & 15) * 2);
    }
#pragma unroll
    for (int i = 0; i < 2; ++i) {
      int q = i * 128 + tid;
      const float* bp = Bw + (size_t)(kc + 2 * (q >> 4)) * N + col0 +
                        (q & 15) * 4;
      fb0[i] = *(const float4*)(bp);
      fb1[i] = *(const float4*)(bp + N);
    }
  };
  auto store_phase = [&](int buf) {
#pragma unroll
    for (int i = 0; i < 8; ++i) {
      int p = i * 128 + tid;
      As[buf][(p >> 4) * AS_STRIDE + (p & 15)] = pk2(fa[i].x, fa[i].y);
    }
#pragma unroll
    for (int i = 0; i < 2; ++i) {
      int q = i * 128 + tid;
      int kp = q >> 4, cc = q & 15;
      Bs[buf][(cc * 4 + 0) * BS_STRIDE + kp] = pk2(fb0[i].x, fb1[i].x);
      Bs[buf][(cc * 4 + 1) * BS_STRIDE + kp] = pk2(fb0[i].y, fb1[i].y);
      Bs[buf][(cc * 4 + 2) * BS_STRIDE + kp] = pk2(fb0[i].z, fb1[i].z);
      Bs[buf][(cc * 4 + 3) * BS_STRIDE + kp] = pk2(fb0[i].w, fb1[i].w);
    }
  };

  v8f acc[4];
  const v8f zf = {0.f, 0.f, 0.f, 0.f, 0.f, 0.f, 0.f, 0.f};
#pragma unroll
  for (int c = 0; c < 4; ++c) acc[c] = zf;

  load_phase(0);
  store_phase(0);
  __syncthreads();

  int buf = 0;
  for (int kc = 0; kc < K; kc += 32) {
    const bool has_next = (kc + 32 < K);
    if (has_next) load_phase(kc + 32);   // global loads fly during WMMAs

    const unsigned int* arow = &As[buf][(wv * 16 + l15) * AS_STRIDE];
    v8u af = mk8(*(const uint4*)(arow + 8 * half),
                 *(const uint4*)(arow + 8 * half + 4));
    v8u bfv[4];
#pragma unroll
    for (int c = 0; c < 4; ++c) {
      const unsigned int* bcol = &Bs[buf][(c * 16 + l15) * BS_STRIDE + 8 * half];
      bfv[c] = mk8(*(const uint4*)(bcol), *(const uint4*)(bcol + 4));
    }
#pragma unroll
    for (int c = 0; c < 4; ++c) acc[c] = wmma_bf16(af, bfv[c], acc[c]);

    if (has_next) store_phase(buf ^ 1);
    __syncthreads();
    buf ^= 1;
  }

  // C/D layout: VGPR j -> row (half*8 + j), col = l15; offsets fold to imms.
  float* crow = C + (size_t)(row0 + wv * 16 + half * 8) * N + col0 + l15;
#pragma unroll
  for (int j = 0; j < 8; ++j)
#pragma unroll
    for (int c = 0; c < 4; ++c)
      crow[(size_t)j * N + c * 16] = acc[c][j];
}

// ---------------------------------------------------------------------------
// RoPE applied in place to q and k; rotated K also emitted as bf16 tiles
// (layout [b*NKVH+kvh][t][HDIM]) for async staging in the attention kernel.
// ---------------------------------------------------------------------------
__global__ void __launch_bounds__(256)
rope_kernel(float* __restrict__ qkv, unsigned short* __restrict__ kbf,
            const int* __restrict__ positions) {
  int idx = blockIdx.x * 256 + threadIdx.x;
  int i  = idx & 127;
  int r  = idx >> 7;
  int h  = r % 12;               // 0-7: q heads, 8-11: k heads
  int bt = r / 12;               // 0 .. B*T-1

  float fraction = (float)i * (1.0f / 128.0f);
  float inv_ts   = exp2f(-fraction * ROPE_LOG2_BASE);
  float ang      = (float)positions[bt] * inv_ts;
  float s, c;
  __sincosf(ang, &s, &c);

  size_t base = (size_t)bt * QKVD + (h < 8 ? h * HDIM : KOFF + (h - 8) * HDIM);
  float x1 = qkv[base + i];
  float x2 = qkv[base + 128 + i];
  float y1 = x1 * c - x2 * s;
  float y2 = x2 * c + x1 * s;
  qkv[base + i]       = y1;
  qkv[base + 128 + i] = y2;

  if (h >= 8) {                   // bf16 K copy for the TDM/async path
    int b_ = bt >> 11, t_ = bt & (TT - 1);
    size_t kb = ((size_t)(b_ * NKVH + (h - 8)) * TT + t_) * HDIM;
    kbf[kb + i]       = f2bf(y1);
    kbf[kb + 128 + i] = f2bf(y2);
  }
}

// ---------------------------------------------------------------------------
// Pack V as bf16 V^T: vbfT[b*NKVH+kvh][dim][t]  (key axis contiguous, so
// attention PV B-fragments are raw 16B-copyable).  LDS tile transpose.
// ---------------------------------------------------------------------------
__global__ void __launch_bounds__(256)
vpack_kernel(const float* __restrict__ qkv, unsigned short* __restrict__ vbfT) {
  __shared__ __align__(16) unsigned short tile[64][68];   // [dim][t], padded
  const int tb  = blockIdx.x;
  const int t0  = (tb & 31) * 64;
  const int d0  = ((tb >> 5) & 3) * 64;
  const int bh  = tb >> 7;               // b*NKVH + kvh
  const int b_  = bh >> 2, kvh = bh & 3;
  const int tid = threadIdx.x;

#pragma unroll
  for (int i = 0; i < 4; ++i) {
    int e = tid + i * 256;
    int tt = e >> 4, dd4 = (e & 15) * 4;
    float4 f = *(const float4*)(qkv + (size_t)(b_ * TT + t0 + tt) * QKVD +
                                VOFF + kvh * HDIM + d0 + dd4);
    tile[dd4 + 0][tt] = f2bf(f.x);
    tile[dd4 + 1][tt] = f2bf(f.y);
    tile[dd4 + 2][tt] = f2bf(f.z);
    tile[dd4 + 3][tt] = f2bf(f.w);
  }
  __syncthreads();
#pragma unroll
  for (int i = 0; i < 4; ++i) {
    int e = tid + i * 256;
    int dd = e >> 4, tc = (e & 15) * 4;
    uint2 v = *(const uint2*)&tile[dd][tc];
    *(uint2*)(vbfT + ((size_t)bh * HDIM + d0 + dd) * TT + t0 + tc) = v;
  }
}

// ---------------------------------------------------------------------------
// Flash attention with tanh soft-cap. Block = 64 threads (2 waves).
// K/V tiles DMA'd with GLOBAL_LOAD_ASYNC_TO_LDS_B128 into double-buffered
// LDS (wave 0: K tile, wave 1: V^T tile); chunk n+1 copies overlap chunk n
// compute; ASYNCcnt (in-order) + barrier separate producer/consumer.
// ---------------------------------------------------------------------------
#define KT_BYTES 528    // K tile row: 512B used + 16B pad (16B-aligned)
#define VT_BYTES 80     // V^T tile row: 64B used + 16B pad
#define PL_STRIDE 40    // ushort per P row (32 used + pad), 80B = 16B-aligned

__global__ void __launch_bounds__(64)
attn_kernel(const float* __restrict__ qkv,
            const unsigned short* __restrict__ kbf,
            const unsigned short* __restrict__ vbfT,
            float* __restrict__ enc) {
  __shared__ __align__(16) unsigned char  ktile[2][32 * KT_BYTES];
  __shared__ __align__(16) unsigned char  vtile[2][256 * VT_BYTES];
  __shared__ __align__(16) unsigned short pl[2 * 16 * PL_STRIDE];

  const int tid  = threadIdx.x;
  const int w    = tid >> 5;       // wave id == query group g
  const int lane = tid & 31;
  const int l15  = lane & 15;
  const int half = lane >> 4;

  const int bid = blockIdx.x;
  const int qt  = bid & 127;          // query tile (16 rows)
  const int kvh = (bid >> 7) & 3;     // kv head
  const int b   = bid >> 9;           // batch
  const int h   = kvh * 2 + w;        // full query head index

  const unsigned short* kg = kbf + (size_t)(b * NKVH + kvh) * TT * HDIM;
  const unsigned short* vg = vbfT + (size_t)(b * NKVH + kvh) * HDIM * TT;

  // Issue async memory->LDS copies for one 32-key chunk (no wait here).
  auto issue_copy = [&](int bufi, int s0c) {
    if (w == 0) {        // K tile: 32 rows x 512B, 32 lanes x 16B per op
      unsigned lbase = (unsigned)(size_t)(void*)&ktile[bufi][0] +
                       (unsigned)(lane * 16);
      unsigned long long g =
          (unsigned long long)(size_t)(kg + (size_t)s0c * HDIM) + lane * 16;
#pragma unroll
      for (int u = 0; u < 32; ++u) {
        unsigned lds = lbase + u * KT_BYTES;
        unsigned long long ga = g + (unsigned long long)u * (HDIM * 2);
        asm volatile("global_load_async_to_lds_b128 %0, %1, off"
                     :: "v"(lds), "v"(ga) : "memory");
      }
    } else {             // V^T tile: 256 rows x 64B; lane = (row%8, 16B chunk)
      int dr = lane >> 2, c = lane & 3;
      unsigned lbase = (unsigned)(size_t)(void*)&vtile[bufi][0] +
                       (unsigned)(dr * VT_BYTES + c * 16);
      unsigned long long g =
          (unsigned long long)(size_t)(vg + (size_t)dr * TT + s0c) + c * 16;
#pragma unroll
      for (int u = 0; u < 32; ++u) {
        unsigned lds = lbase + u * (8 * VT_BYTES);
        unsigned long long ga = g + (unsigned long long)u * (8 * TT * 2);
        asm volatile("global_load_async_to_lds_b128 %0, %1, off"
                     :: "v"(lds), "v"(ga) : "memory");
      }
    }
  };

  // ---- Preload Q fragments: 8 chunks of (16x32) A-layout bf16 ----
  v8u qf[8];
  {
    int trow = qt * 16 + l15;
    const float* qp = qkv + (size_t)(b * TT + trow) * QKVD + h * HDIM;
#pragma unroll
    for (int kc = 0; kc < 8; ++kc) {
      int k0 = kc * 32 + 16 * half;       // j=0..3 -> dims k0..k0+7
      int k1 = k0 + 8;                    // j=4..7 -> dims k1..k1+7
      float4 q0 = *(const float4*)(qp + k0);
      float4 q1 = *(const float4*)(qp + k0 + 4);
      float4 q2 = *(const float4*)(qp + k1);
      float4 q3 = *(const float4*)(qp + k1 + 4);
      qf[kc][0] = pk2(q0.x, q0.y); qf[kc][1] = pk2(q0.z, q0.w);
      qf[kc][2] = pk2(q1.x, q1.y); qf[kc][3] = pk2(q1.z, q1.w);
      qf[kc][4] = pk2(q2.x, q2.y); qf[kc][5] = pk2(q2.z, q2.w);
      qf[kc][6] = pk2(q3.x, q3.y); qf[kc][7] = pk2(q3.z, q3.w);
    }
  }

  const v8f zf = {0.f, 0.f, 0.f, 0.f, 0.f, 0.f, 0.f, 0.f};
  v8f o[16];                        // 16x256 fp32 output accumulator
#pragma unroll
  for (int c = 0; c < 16; ++c) o[c] = zf;
  float mrun[8], lrun[8];
#pragma unroll
  for (int j = 0; j < 8; ++j) { mrun[j] = -INFINITY; lrun[j] = 0.f; }

  const int nchunks = (qt * 16 + 16 + 31) >> 5;   // causal: keys <= qt*16+15

  issue_copy(0, 0);
  int buf = 0;
  for (int sc = 0; sc < nchunks; ++sc) {
    const int s0 = sc * 32;
    const bool more = (sc + 1 < nchunks);

    if (more) issue_copy(buf ^ 1, s0 + 32);   // overlap with this chunk
    // In-order ASYNCcnt: <=32 outstanding means this chunk's copies landed.
    if (more) asm volatile("s_wait_asynccnt 0x20" ::: "memory");
    else      asm volatile("s_wait_asynccnt 0x0"  ::: "memory");
    __syncthreads();

    // ---- S = Q * K^T : two 16x16 logit tiles; 2 k-chunks per load group ----
    v8f sa = zf, sb = zf;
#pragma unroll
    for (int kc = 0; kc < 8; kc += 2) {
      const unsigned int* k0p =
          (const unsigned int*)(ktile[buf] + l15 * KT_BYTES) + kc * 16 + 8 * half;
      const unsigned int* k1p =
          (const unsigned int*)(ktile[buf] + (16 + l15) * KT_BYTES) + kc * 16 + 8 * half;
      v8u b0a = mk8(*(const uint4*)(k0p), *(const uint4*)(k0p + 4));
      v8u b1a = mk8(*(const uint4*)(k1p), *(const uint4*)(k1p + 4));
      v8u b0b = mk8(*(const uint4*)(k0p + 16), *(const uint4*)(k0p + 20));
      v8u b1b = mk8(*(const uint4*)(k1p + 16), *(const uint4*)(k1p + 20));
      sa = wmma_bf16(qf[kc], b0a, sa);
      sb = wmma_bf16(qf[kc], b1a, sb);
      sa = wmma_bf16(qf[kc + 1], b0b, sa);
      sb = wmma_bf16(qf[kc + 1], b1b, sb);
    }

    // ---- soft-cap, scale, causal mask, online softmax (DPP reductions) ----
    const int qbase = qt * 16 + half * 8;
#pragma unroll
    for (int j = 0; j < 8; ++j) {
      int qr = qbase + j;
      float a0 = fast_tanh(sa[j] * (1.0f / 50.0f)) * (50.0f / 16.0f);
      float a1 = fast_tanh(sb[j] * (1.0f / 50.0f)) * (50.0f / 16.0f);
      a0 = (s0 + l15      <= qr) ? a0 : -INFINITY;
      a1 = (s0 + 16 + l15 <= qr) ? a1 : -INFINITY;

      float cm = red16_max(fmaxf(a0, a1));
      float mn = fmaxf(mrun[j], cm);
      float scale = __expf(mrun[j] - mn);
      float p0 = __expf(a0 - mn);
      float p1 = __expf(a1 - mn);
      float rs = red16_sum(p0 + p1);
      lrun[j] = lrun[j] * scale + rs;
      mrun[j] = mn;

      pl[(w * 16 + half * 8 + j) * PL_STRIDE + l15]      = f2bf(p0);
      pl[(w * 16 + half * 8 + j) * PL_STRIDE + 16 + l15] = f2bf(p1);
#pragma unroll
      for (int c = 0; c < 16; ++c) o[c][j] *= scale;
    }
    asm volatile("" ::: "memory");   // keep P stores before P frag loads

    // ---- O += P (16x32) * V (32x256); 4 tiles per load group ----
    const unsigned int* prow =
        (const unsigned int*)pl + (w * 16 + l15) * (PL_STRIDE / 2);
    v8u pa = mk8(*(const uint4*)(prow + 8 * half),
                 *(const uint4*)(prow + 8 * half + 4));

#pragma unroll
    for (int c0 = 0; c0 < 16; c0 += 4) {
      v8u vb[4];
#pragma unroll
      for (int c = 0; c < 4; ++c) {
        const unsigned int* vrow =
            (const unsigned int*)(vtile[buf] + ((c0 + c) * 16 + l15) * VT_BYTES) +
            8 * half;
        vb[c] = mk8(*(const uint4*)(vrow), *(const uint4*)(vrow + 4));
      }
#pragma unroll
      for (int c = 0; c < 4; ++c)
        o[c0 + c] = wmma_bf16(pa, vb[c], o[c0 + c]);
    }
    __syncthreads();   // all reads of buf done before async overwrites it
    buf ^= 1;
  }

  // ---- normalize and store encoded[b, t, h, :] ----
#pragma unroll
  for (int j = 0; j < 8; ++j) {
    float inv = 1.0f / lrun[j];
    int t = qt * 16 + half * 8 + j;
    float* op = enc + ((size_t)(b * TT + t) * NHEAD + h) * HDIM;
#pragma unroll
    for (int c = 0; c < 16; ++c) op[c * 16 + l15] = o[c][j] * inv;
  }
}

// ---------------------------------------------------------------------------
extern "C" void kernel_launch(void* const* d_in, const int* in_sizes, int n_in,
                              void* d_out, int out_size, void* d_ws,
                              size_t ws_size, hipStream_t stream) {
  (void)in_sizes; (void)n_in; (void)out_size; (void)ws_size;
  const float* x         = (const float*)d_in[0];
  const int*   positions = (const int*)d_in[1];
  // d_in[2] = attn_mask (causal; applied analytically in attn_kernel)
  const float* w_qkv     = (const float*)d_in[3];
  const float* w_out     = (const float*)d_in[4];
  float*       out       = (float*)d_out;

  float* qkv = (float*)d_ws;                              // [B*T, 4096] f32
  float* enc = qkv + (size_t)BB * TT * QKVD;              // [B*T, 2048] f32
  unsigned short* kbf  = (unsigned short*)(enc + (size_t)BB * TT * DD);
  unsigned short* vbfT = kbf + (size_t)BB * NKVH * TT * HDIM;

  const int M = BB * TT;  // 8192

  // 1) qkv = x @ w_qkv
  gemm_kernel<QKVD, DD><<<dim3(QKVD / 64, M / 64), 128, 0, stream>>>(
      x, w_qkv, qkv);

  // 2) RoPE on q and k; also emits bf16 K tiles
  rope_kernel<<<(BB * TT * 12 * 128) / 256, 256, 0, stream>>>(qkv, kbf,
                                                              positions);

  // 3) bf16 V^T tiles
  vpack_kernel<<<BB * NKVH * (TT / 64) * (HDIM / 64), 256, 0, stream>>>(qkv,
                                                                        vbfT);

  // 4) grouped causal flash attention with tanh soft-cap (async-LDS staging)
  attn_kernel<<<BB * NKVH * (TT / 16), 64, 0, stream>>>(qkv, kbf, vbfT, enc);

  // 5) out = encoded @ w_out
  gemm_kernel<DD, DD><<<dim3(DD / 64, M / 64), 128, 0, stream>>>(
      enc, w_out, out);
}
